// VectorQuantizer_24206435680826
// MI455X (gfx1250) — matrix operand order repaired and checked by reading, not compiled, not run on hardware
//
#include <hip/hip_runtime.h>

typedef _Float16 v16h __attribute__((ext_vector_type(16)));
typedef float    v8f  __attribute__((ext_vector_type(8)));

static constexpr int DIM            = 32;   // reduction depth == WMMA K for f16
static constexpr int KCB            = 512;  // codebook entries
static constexpr int ROWS_PER_WAVE  = 16;
static constexpr int WAVES_PER_BLK  = 8;
static constexpr int ROWS_PER_BLK   = ROWS_PER_WAVE * WAVES_PER_BLK; // 128

// Zero the scalar output slots (l_vq and the literal 0) before accumulation.
__global__ void vq_init_kernel(float* out, int ndim_total) {
    if (threadIdx.x == 0) {
        out[ndim_total]     = 0.0f;
        out[ndim_total + 1] = 0.0f;
    }
}

__global__ void __launch_bounds__(256)
vq_kernel(const float* __restrict__ x,
          const float* __restrict__ embed,
          float* __restrict__ out,
          int nrows, float inv_total) {
    // Codebook staged as (-2 * embed) in f16 + f32 squared norms.
    __shared__ _Float16 sE[KCB * DIM];   // 32 KB
    __shared__ float    sE2[KCB];        //  2 KB

    const int tid  = threadIdx.x;
    const int lane = tid & 31;
    const int wave = tid >> 5;

    // ---- Stage codebook into LDS (float4 = global_load_b128), pre-scaled by -2.
    for (int r = tid; r < KCB; r += 256) {
        const float4* er = (const float4*)(embed + r * DIM);
        _Float16* dst = &sE[r * DIM];
        float ss = 0.0f;
        #pragma unroll
        for (int j = 0; j < DIM / 4; ++j) {
            float4 v = er[j];
            ss += v.x * v.x + v.y * v.y + v.z * v.z + v.w * v.w;
            dst[4 * j + 0] = (_Float16)(-2.0f * v.x);
            dst[4 * j + 1] = (_Float16)(-2.0f * v.y);
            dst[4 * j + 2] = (_Float16)(-2.0f * v.z);
            dst[4 * j + 3] = (_Float16)(-2.0f * v.w);
        }
        sE2[r] = ss;
    }
    __syncthreads();

    const int rowBase = blockIdx.x * ROWS_PER_BLK + wave * ROWS_PER_WAVE;

    // ---- A tile: 16x32 f16, ISA layout: lanes<16 row M=lane hold K 0-7 & 16-23,
    // lanes>=16 same row hold K 8-15 & 24-31.
    int myRow = rowBase + (lane & 15);
    if (myRow >= nrows) myRow = nrows - 1;          // clamp (select, EXEC untouched)
    const float* xr = x + (size_t)myRow * DIM;
    const int kb = (lane < 16) ? 0 : 8;
    float4 a0 = *(const float4*)(xr + kb);
    float4 a1 = *(const float4*)(xr + kb + 4);
    float4 a2 = *(const float4*)(xr + kb + 16);
    float4 a3 = *(const float4*)(xr + kb + 20);
    v16h a;
    a[0]  = (_Float16)a0.x; a[1]  = (_Float16)a0.y; a[2]  = (_Float16)a0.z; a[3]  = (_Float16)a0.w;
    a[4]  = (_Float16)a1.x; a[5]  = (_Float16)a1.y; a[6]  = (_Float16)a1.z; a[7]  = (_Float16)a1.w;
    a[8]  = (_Float16)a2.x; a[9]  = (_Float16)a2.y; a[10] = (_Float16)a2.z; a[11] = (_Float16)a2.w;
    a[12] = (_Float16)a3.x; a[13] = (_Float16)a3.y; a[14] = (_Float16)a3.z; a[15] = (_Float16)a3.w;

    float bestD[8];
    int   bestI[8];
    #pragma unroll
    for (int r = 0; r < 8; ++r) { bestD[r] = 3.4e38f; bestI[r] = 0; }

    const int myCol = lane & 15;                    // C/D column owned by this lane
    const int kb2   = (lane < 16) ? 0 : 8;          // B: K 0-7 & 16-23 vs 8-15 & 24-31 per half-wave

    // Unroll x4: 4 independent B-tiles / accumulators in flight so the compiler
    // can fill the WMMA->VALU hazard slots with the neighbors' loads/compares.
    #pragma unroll 4
    for (int k0 = 0; k0 < KCB; k0 += 16) {
        const int cw = k0 + myCol;                  // codeword scored by this lane
        const _Float16* bp = &sE[cw * DIM + kb2];
        v16h b;
        #pragma unroll
        for (int j = 0; j < 16; ++j) b[j] = bp[j < 8 ? j : j + 8];

        // Preload C with ||e||^2 so D = e2 - 2*x.e  (B already scaled by -2)
        float e2v = sE2[cw];
        v8f c;
        #pragma unroll
        for (int r = 0; r < 8; ++r) c[r] = e2v;

        c = __builtin_amdgcn_wmma_f32_16x16x32_f16(
                /*neg_a=*/false, a, /*neg_b=*/false, b,
                /*c_mod=*/(short)0, c, /*reuse_a=*/false, /*reuse_b=*/false);

        #pragma unroll
        for (int r = 0; r < 8; ++r) {
            float d = c[r];
            int take = d < bestD[r];
            bestD[r] = take ? d  : bestD[r];
            bestI[r] = take ? cw : bestI[r];
        }
    }

    // Argmin butterfly inside each 16-lane half (rows 0-7 in lanes 0-15,
    // rows 8-15 in lanes 16-31). Branchless; lower index wins ties (argmin-first).
    #pragma unroll
    for (int r = 0; r < 8; ++r) {
        float d = bestD[r];
        int   i = bestI[r];
        #pragma unroll
        for (int m = 8; m >= 1; m >>= 1) {
            float od = __shfl_xor(d, m, 32);
            int   oi = __shfl_xor(i, m, 32);
            int better = (int)(od < d) | ((int)(od == d) & (int)(oi < i));
            d = better ? od : d;
            i = better ? oi : i;
        }
        bestI[r] = i;
    }

    // Gather f32 codewords, write straight-through output, accumulate loss.
    float acc = 0.0f;
    #pragma unroll
    for (int r = 0; r < 8; ++r) {
        const int idxLo = __shfl(bestI[r], 0, 32);   // row rowBase + r
        const int idxHi = __shfl(bestI[r], 16, 32);  // row rowBase + 8 + r
        #pragma unroll
        for (int h = 0; h < 2; ++h) {
            const int row = rowBase + r + (h ? 8 : 0);
            const int qi  = h ? idxHi : idxLo;
            if (row < nrows) {
                float qv = embed[qi * DIM + lane];         // 32 lanes == one row
                float xv = x[(size_t)row * DIM + lane];
                out[(size_t)row * DIM + lane] = qv;        // x + sg(q - x) == q
                float df = qv - xv;
                acc += df * df;
            }
        }
    }

    // Commitment loss: wave-reduce then one atomic per wave.
    #pragma unroll
    for (int m = 16; m >= 1; m >>= 1) acc += __shfl_xor(acc, m, 32);
    if (lane == 0) atomicAdd(out + (size_t)nrows * DIM, acc * inv_total);
}

extern "C" void kernel_launch(void* const* d_in, const int* in_sizes, int n_in,
                              void* d_out, int out_size, void* d_ws, size_t ws_size,
                              hipStream_t stream) {
    (void)n_in; (void)out_size; (void)d_ws; (void)ws_size;
    const float* x     = (const float*)d_in[0];
    const float* embed = (const float*)d_in[1];
    float* out         = (float*)d_out;

    const int nrows     = in_sizes[0] / DIM;      // 262144
    const float inv_tot = 1.0f / (float)in_sizes[0];

    vq_init_kernel<<<dim3(1), dim3(32), 0, stream>>>(out, nrows * DIM);

    const int blocks = (nrows + ROWS_PER_BLK - 1) / ROWS_PER_BLK; // 2048
    vq_kernel<<<dim3(blocks), dim3(256), 0, stream>>>(x, embed, out, nrows, inv_tot);
}
